// ONLSTM_747324310360
// MI455X (gfx1250) — compile-verified, matrix operand order checked by reading, not run on hardware
//
#include <hip/hip_runtime.h>
#include <stdint.h>

// ---------------------------------------------------------------------------
// ONLSTM (B=32,S=512,H=512,L=2) persistent-scan kernel for MI455X (gfx1250).
//  - bf16 WMMA 16x16x32, f32 accumulate; A/B operands pre-swizzled to the
//    exact wave32 VGPR fragment layout from CDNA5 ISA 7.12.2.
//  - 17 workgroups x 256 threads (136 waves) persistent; 2 grid barriers/step.
//  - Weights converted once to bf16 fragments (8.9MB, L2-resident at 192MB).
// ---------------------------------------------------------------------------

typedef __attribute__((ext_vector_type(16))) __bf16 v16bf;
typedef __attribute__((ext_vector_type(8)))  float  v8f;

#define B_      32
#define S_      512
#define H_      512
#define L_      2
#define CS_     64
#define G_      2176          // 2*CS + 4*H
#define NSTRIP  136           // G/16
#define NWG     17
#define WPB     8             // waves per workgroup (wave32)
#define TPB     256

// A-panel for one step: 16 chunks * 2 mtiles * 32 lanes * 16 elems = 16384 bf16
#define PANEL_ELEMS 16384u

// workspace layout (bytes, all 256-aligned)
#define BAR_OFF    0u
#define GATES_OFF  256u                                  // 32*2176 f32
#define CBUF_OFF   (GATES_OFF + (unsigned)(B_*G_*4))     // 32*512 f32
#define HPAN_OFF   (CBUF_OFF  + (unsigned)(B_*H_*4))     // 2 parities * PANEL * 2B
#define WFRAG_OFF  (HPAN_OFF  + 2u*PANEL_ELEMS*2u)       // L*136*32ch*512 * 2B
#define XP0_OFF    (WFRAG_OFF + (unsigned)(L_*NSTRIP*32*512*2))
#define XP1_OFF    (XP0_OFF   + (unsigned)S_*PANEL_ELEMS*2u)
#define WS_BYTES   (XP1_OFF   + (unsigned)S_*PANEL_ELEMS*2u)   // ~42.9 MB

__device__ __forceinline__ unsigned short f2bf(float f) {
  union { float f; unsigned u; } v; v.f = f;
  unsigned u = v.u;
  u += 0x7FFFu + ((u >> 16) & 1u);          // round-to-nearest-even
  return (unsigned short)(u >> 16);
}

// ISA 7.12.2 16-bit fragment K-mapping: element i of lane -> K offset in chunk
__device__ __forceinline__ int kmap(int lane, int i) {
  return i + ((lane >> 4) << 3) + ((i >= 8) ? 8 : 0);
}

__device__ __forceinline__ void grid_sync(unsigned* bar) {
  __threadfence();
  __syncthreads();
  if (threadIdx.x == 0) {
    volatile unsigned* vgen = bar + 1;
    unsigned g = *vgen;
    if (atomicAdd(bar, 1u) == NWG - 1) {
      bar[0] = 0;
      __threadfence();
      atomicAdd((unsigned*)(bar + 1), 1u);
    } else {
      while (*vgen == g) __builtin_amdgcn_s_sleep(1);
    }
  }
  __syncthreads();
}

// inclusive cumsoftmax over 64 values held as (a0,a1) = idx (2*lane, 2*lane+1)
__device__ __forceinline__ void cumsoftmax64(float a0, float a1, int lane,
                                             float& c0, float& c1) {
  float m = fmaxf(a0, a1);
  #pragma unroll
  for (int d = 16; d > 0; d >>= 1) m = fmaxf(m, __shfl_xor(m, d, 32));
  float e0 = __expf(a0 - m), e1 = __expf(a1 - m);
  float run = e0 + e1;
  #pragma unroll
  for (int d = 1; d < 32; d <<= 1) {
    float v = __shfl_up(run, d, 32);
    if (lane >= d) run += v;
  }
  float total = __shfl(run, 31, 32);
  c1 = run / total;
  c0 = (run - e1) / total;
}

__device__ __forceinline__ float sigf(float x) {
  return 1.0f / (1.0f + __expf(-x));
}

// ---------------------------------------------------------------------------
// Prep: concat-weight [W_ih; W_hh]^T (1024 x 2176) -> bf16 B-fragments
// layout: [(l*136+s)][chunk 0..31][lane 0..31][i 0..15]
// ---------------------------------------------------------------------------
__global__ void prep_wfrag(const float* __restrict__ W_ih,
                           const float* __restrict__ W_hh,
                           unsigned short* __restrict__ wfrag) {
  int idx = blockIdx.x * TPB + threadIdx.x;
  if (idx >= L_ * NSTRIP * 32 * 32 * 16) return;
  int i    = idx & 15;
  int lane = (idx >> 4) & 31;
  int ch   = (idx >> 9) & 31;
  int rest = idx >> 14;                 // l*NSTRIP + s
  int s = rest % NSTRIP;
  int l = rest / NSTRIP;
  int k = ch * 32 + kmap(lane, i);
  int n = s * 16 + (lane & 15);
  float v = (k < H_) ? W_ih[((size_t)l * G_ + n) * H_ + k]
                     : W_hh[((size_t)l * G_ + n) * H_ + (k - H_)];
  wfrag[idx] = f2bf(v);
}

// ---------------------------------------------------------------------------
// Prep: layer-0 input x (B,S,H) -> per-step bf16 A-panels
// layout: [t][chunk 0..15][mtile 0..1][lane 0..31][i 0..15]
// ---------------------------------------------------------------------------
__global__ void prep_xp0(const float* __restrict__ x,
                         unsigned short* __restrict__ xp0) {
  int idx = blockIdx.x * TPB + threadIdx.x;
  if (idx >= S_ * (int)PANEL_ELEMS) return;
  int i    = idx & 15;
  int lane = (idx >> 4) & 31;
  int mt   = (idx >> 9) & 1;
  int ch   = (idx >> 10) & 15;
  int t    = idx >> 14;
  int col  = ch * 32 + kmap(lane, i);
  int b    = mt * 16 + (lane & 15);
  xp0[idx] = f2bf(x[((size_t)b * S_ + t) * H_ + col]);
}

// ---------------------------------------------------------------------------
// Persistent scan kernel: 17 blocks x 256 threads, resident for both layers.
// ---------------------------------------------------------------------------
__global__ __launch_bounds__(TPB) void onlstm_scan(
    const float* __restrict__ b_ih, const float* __restrict__ b_hh,
    unsigned char* __restrict__ ws, float* __restrict__ out) {
  unsigned*       bar   = (unsigned*)(ws + BAR_OFF);
  float*          gates = (float*)(ws + GATES_OFF);
  float*          cbuf  = (float*)(ws + CBUF_OFF);
  unsigned short* hpan  = (unsigned short*)(ws + HPAN_OFF);
  const unsigned short* wfrag = (const unsigned short*)(ws + WFRAG_OFF);
  const unsigned short* xp0   = (const unsigned short*)(ws + XP0_OFF);
  unsigned short* xp1   = (unsigned short*)(ws + XP1_OFF);

  const int tid  = threadIdx.x;
  const int lane = tid & 31;
  const int wid  = blockIdx.x * WPB + (tid >> 5);   // 0..135
  const int gtid = blockIdx.x * TPB + tid;          // 0..4351

  for (int l = 0; l < L_; ++l) {
    const unsigned short* xcur = (l == 0) ? xp0 : (const unsigned short*)xp1;
    // zero h fragment panels (both parities) and cell state
    for (int i = gtid; i < (int)(2 * PANEL_ELEMS); i += NWG * TPB) hpan[i] = 0;
    for (int i = gtid; i < B_ * H_; i += NWG * TPB) cbuf[i] = 0.0f;
    grid_sync(bar);

    for (int t = 0; t < S_; ++t) {
      // ---------------- GEMM phase: gates_raw = [x_t, h] @ Wcat^T ----------
      {
        const int s = wid;                                   // N-strip
        const unsigned short* wb = wfrag + (size_t)(l * NSTRIP + s) * 16384;
        const unsigned short* xb = xcur + (size_t)t * PANEL_ELEMS;
        const unsigned short* hb = hpan + (size_t)(t & 1) * PANEL_ELEMS;
        v8f acc0 = {};
        v8f acc1 = {};
        #pragma unroll 4
        for (int ch = 0; ch < 32; ++ch) {
          const unsigned short* ab =
              (ch < 16) ? (xb + ch * 1024) : (hb + (ch - 16) * 1024);
          v16bf a0 = *(const v16bf*)(const void*)(ab + lane * 16);
          v16bf a1 = *(const v16bf*)(const void*)(ab + 512 + lane * 16);
          v16bf bm = *(const v16bf*)(const void*)(wb + ch * 512 + lane * 16);
          acc0 = __builtin_amdgcn_wmma_f32_16x16x32_bf16(
              false, a0, false, bm, (short)0, acc0, false, false);
          acc1 = __builtin_amdgcn_wmma_f32_16x16x32_bf16(
              false, a1, false, bm, (short)0, acc1, false, false);
        }
        // C layout: VGPR r, lane -> row = mt*16 + 8*(lane>=16)+r, col = lane&15
        int col = s * 16 + (lane & 15);
        int rb  = (lane >> 4) * 8;
        #pragma unroll
        for (int r = 0; r < 8; ++r) {
          gates[(size_t)(rb + r) * G_ + col]      = acc0[r];
          gates[(size_t)(16 + rb + r) * G_ + col] = acc1[r];
        }
      }
      grid_sync(bar);

      // ---------------- gate/activation phase: one wave per batch row ------
      if (wid < B_) {
        const int b = wid;
        const float* gr = gates + (size_t)b * G_;
        const float* bi = b_ih + (size_t)l * G_;
        const float* bh = b_hh + (size_t)l * G_;
        // cin = 1 - cumsoftmax(g[0:64]); cforget = cumsoftmax(g[64:128])
        float ci0, ci1, cf0, cf1;
        {
          int i0 = 2 * lane, i1 = 2 * lane + 1;
          float a0 = gr[i0] + bi[i0] + bh[i0];
          float a1 = gr[i1] + bi[i1] + bh[i1];
          float c0, c1;
          cumsoftmax64(a0, a1, lane, c0, c1);
          ci0 = 1.0f - c0; ci1 = 1.0f - c1;
          a0 = gr[64 + i0] + bi[64 + i0] + bh[64 + i0];
          a1 = gr[64 + i1] + bi[64 + i1] + bh[64 + i1];
          cumsoftmax64(a0, a1, lane, cf0, cf1);
        }
        unsigned short* hdst = hpan + (size_t)((t + 1) & 1) * PANEL_ELEMS;
        #pragma unroll
        for (int js = 0; js < 2; ++js) {
          int   j  = 2 * lane + js;
          float ci = js ? ci1 : ci0;
          float cf = js ? cf1 : cf0;
          float ov = cf * ci;
          #pragma unroll
          for (int n = 0; n < 8; ++n) {
            int e  = j * 8 + n;                 // 0..511, element of h/c
            int cb = 128 + j * 8 + n;           // base gate column
            float ing = gr[cb]        + bi[cb]        + bh[cb];
            float fgg = gr[cb + 512]  + bi[cb + 512]  + bh[cb + 512];
            float ogg = gr[cb + 1024] + bi[cb + 1024] + bh[cb + 1024];
            float cgg = gr[cb + 1536] + bi[cb + 1536] + bh[cb + 1536];
            float fgate = sigf(fgg) * ov + (cf - ov);
            float igate = sigf(ing) * ov + (ci - ov);
            float cold  = cbuf[(size_t)b * H_ + e];
            float cnew  = fgate * cold + igate * tanhf(cgg);
            float hnew  = sigf(ogg) * tanhf(cnew);
            cbuf[(size_t)b * H_ + e] = cnew;
            // scatter h_next into bf16 A-fragment layout for next step's GEMM
            int ch   = e >> 5, kk = e & 31;
            int hl   = (kk >> 3) & 1;
            int ii   = (kk & 7) + ((kk >= 16) ? 8 : 0);
            int fln  = (b & 15) + 16 * hl;
            int mt   = b >> 4;
            unsigned short hb16 = f2bf(hnew);
            hdst[(size_t)(ch * 2 + mt) * 512 + fln * 16 + ii] = hb16;
            if (l == 0) {  // layer-0 output feeds layer-1 input panel
              xp1[(size_t)t * PANEL_ELEMS + (size_t)(ch * 2 + mt) * 512 +
                  fln * 16 + ii] = hb16;
            } else {       // final layer output (B,S,H)
              out[(size_t)b * S_ * H_ + (size_t)t * H_ + e] = hnew;
            }
            if (t == S_ - 1) {  // hT (L,B,H) then cT (L,B,H)
              size_t base = (size_t)B_ * S_ * H_;
              out[base + ((size_t)l * B_ + b) * H_ + e] = hnew;
              out[base + (size_t)L_ * B_ * H_ + ((size_t)l * B_ + b) * H_ + e] = cnew;
            }
          }
        }
      }
      grid_sync(bar);
    }
  }
}

extern "C" void kernel_launch(void* const* d_in, const int* in_sizes, int n_in,
                              void* d_out, int out_size, void* d_ws,
                              size_t ws_size, hipStream_t stream) {
  const float* x    = (const float*)d_in[0];
  const float* W_ih = (const float*)d_in[1];
  const float* b_ih = (const float*)d_in[2];
  const float* W_hh = (const float*)d_in[3];
  const float* b_hh = (const float*)d_in[4];
  unsigned char* ws = (unsigned char*)d_ws;   // needs WS_BYTES (~43 MB)
  float* out = (float*)d_out;

  hipMemsetAsync(ws + BAR_OFF, 0, 256, stream);

  {
    int total = L_ * NSTRIP * 32 * 32 * 16;
    prep_wfrag<<<(total + TPB - 1) / TPB, TPB, 0, stream>>>(
        W_ih, W_hh, (unsigned short*)(ws + WFRAG_OFF));
  }
  {
    int total = S_ * (int)PANEL_ELEMS;
    prep_xp0<<<(total + TPB - 1) / TPB, TPB, 0, stream>>>(
        x, (unsigned short*)(ws + XP0_OFF));
  }
  onlstm_scan<<<NWG, TPB, 0, stream>>>(b_ih, b_hh, ws, out);
}
